// WalkerVisitLosses_13443247636928
// MI455X (gfx1250) — matrix operand order repaired and checked by reading, not compiled, not run on hardware
//
#include <hip/hip_runtime.h>

// ---------------------------------------------------------------------------
// WalkerVisitLosses on MI455X (gfx1250, wave32, WMMA bf16)
// N = M = 8192, D = 128, NUM_LABELS = 10
// ---------------------------------------------------------------------------

#define NN 8192
#define MM 8192
#define DD 128
#define NLAB 10
#define EPS 1e-8f

typedef __attribute__((ext_vector_type(16))) __bf16 v16bf;
typedef __attribute__((ext_vector_type(8)))  float  v8f;

union FragBf {
    v16bf v;
    uint4 q[2];
};

__device__ __forceinline__ v8f wmma_bf16(const FragBf& a, const FragBf& b, v8f c) {
    return __builtin_amdgcn_wmma_f32_16x16x32_bf16(
        /*neg_a=*/false, a.v, /*neg_b=*/false, b.v,
        /*c_mod=*/(short)0, c, /*reuse_a=*/false, /*reuse_b=*/false);
}

__device__ __forceinline__ unsigned short f32_to_bf16(float f) {
    unsigned int u = __float_as_uint(f);
    unsigned int r = (u + 0x7FFFu + ((u >> 16) & 1u)) >> 16;
    return (unsigned short)r;
}

__device__ __forceinline__ void store_bf16x4(unsigned short* p, float4 f) {
    unsigned int lo = ((unsigned int)f32_to_bf16(f.y) << 16) | (unsigned int)f32_to_bf16(f.x);
    unsigned int hi = ((unsigned int)f32_to_bf16(f.w) << 16) | (unsigned int)f32_to_bf16(f.z);
    uint2 u; u.x = lo; u.y = hi;
    *(uint2*)p = u;
}

// ---------------------------------------------------------------------------
// K0: zero out[0..1] and visitSum[8192]
// ---------------------------------------------------------------------------
__global__ __launch_bounds__(256) void init_kernel(float* out, float* visitSum) {
    int t = blockIdx.x * 256 + threadIdx.x;
    if (t < 2) out[t] = 0.0f;
    if (t < MM) visitSum[t] = 0.0f;
}

// ---------------------------------------------------------------------------
// K1: p = a @ b^T * rsqrt(D).  128x128 output tile per block, KT=64, bf16 WMMA.
// 8 waves: 2 (M) x 4 (N); each wave owns 64 rows x 32 cols = 4x2 WMMA subtiles.
// ---------------------------------------------------------------------------
#define PLDT 72   // 64 + 8 pad, elements (row stride in LDS)

__global__ __launch_bounds__(256) void gemm_p_kernel(const float* __restrict__ A,
                                                     const float* __restrict__ B,
                                                     float* __restrict__ P) {
    __shared__ __align__(16) unsigned short As[128 * PLDT];
    __shared__ __align__(16) unsigned short Bs[128 * PLDT];

    const int t    = threadIdx.x;
    const int lane = t & 31;
    const int wave = t >> 5;
    const int wm   = wave & 1;   // 2 wave-rows of 64
    const int wn   = wave >> 1;  // 4 wave-cols of 32
    const int half = lane >> 4;
    const int l15  = lane & 15;

    const int rowBase = blockIdx.y * 128;
    const int colBase = blockIdx.x * 128;

    v8f acc[4][2];
#pragma unroll
    for (int i = 0; i < 4; i++)
#pragma unroll
        for (int j = 0; j < 2; j++) acc[i][j] = {};

    for (int k0 = 0; k0 < DD; k0 += 64) {
        // stage 128x64 f32 of A and B into LDS as bf16
#pragma unroll
        for (int s = t; s < 128 * 16; s += 256) {
            int r  = s >> 4;
            int c4 = s & 15;
            float4 fa = *(const float4*)(&A[(size_t)(rowBase + r) * DD + k0 + c4 * 4]);
            store_bf16x4(&As[r * PLDT + c4 * 4], fa);
            float4 fb = *(const float4*)(&B[(size_t)(colBase + r) * DD + k0 + c4 * 4]);
            store_bf16x4(&Bs[r * PLDT + c4 * 4], fb);
        }
        __syncthreads();

#pragma unroll
        for (int ks = 0; ks < 64; ks += 32) {
            FragBf aF[4], bF[2];
#pragma unroll
            for (int i = 0; i < 4; i++) {
                const unsigned short* rp = &As[(wm * 64 + i * 16 + l15) * PLDT + ks];
                aF[i].q[0] = *(const uint4*)(rp + 16 * half);
                aF[i].q[1] = *(const uint4*)(rp + 16 * half + 8);
            }
#pragma unroll
            for (int j = 0; j < 2; j++) {
                const unsigned short* rp = &Bs[(wn * 32 + j * 16 + l15) * PLDT + ks];
                bF[j].q[0] = *(const uint4*)(rp + 16 * half);
                bF[j].q[1] = *(const uint4*)(rp + 16 * half + 8);
            }
#pragma unroll
            for (int i = 0; i < 4; i++)
#pragma unroll
                for (int j = 0; j < 2; j++)
                    acc[i][j] = wmma_bf16(aF[i], bF[j], acc[i][j]);
        }
        __syncthreads();
    }

    const float scale = 0.08838834764831843f;  // 1/sqrt(128)
#pragma unroll
    for (int i = 0; i < 4; i++) {
#pragma unroll
        for (int j = 0; j < 2; j++) {
#pragma unroll
            for (int r = 0; r < 8; r++) {
                int row = rowBase + wm * 64 + i * 16 + r + 8 * half;
                int col = colBase + wn * 32 + j * 16 + l15;
                P[(size_t)row * MM + col] = acc[i][j][r] * scale;
            }
        }
    }
}

// ---------------------------------------------------------------------------
// K2: per-row max and 1/sumexp.  One block per row.
// ---------------------------------------------------------------------------
__global__ __launch_bounds__(256) void row_stats_kernel(const float* __restrict__ P,
                                                        float* __restrict__ rowMax,
                                                        float* __restrict__ rowRcp) {
    const int i = blockIdx.x;
    const float* row = &P[(size_t)i * MM];
    __shared__ float red[256];
    const int t = threadIdx.x;

    float m = -3.402823466e+38f;
    for (int k = t; k < MM; k += 256) m = fmaxf(m, row[k]);
    red[t] = m; __syncthreads();
    for (int s = 128; s > 0; s >>= 1) { if (t < s) red[t] = fmaxf(red[t], red[t + s]); __syncthreads(); }
    m = red[0]; __syncthreads();

    float sum = 0.0f;
    for (int k = t; k < MM; k += 256) sum += __expf(row[k] - m);
    red[t] = sum; __syncthreads();
    for (int s = 128; s > 0; s >>= 1) { if (t < s) red[t] += red[t + s]; __syncthreads(); }
    if (t == 0) { rowMax[i] = m; rowRcp[i] = 1.0f / red[0]; }
}

// ---------------------------------------------------------------------------
// K3/K4: per-column max and sumexp, split over 16 row slices then combined.
// ---------------------------------------------------------------------------
__global__ __launch_bounds__(256) void col_part_kernel(const float* __restrict__ P,
                                                       float* __restrict__ pMax,
                                                       float* __restrict__ pSum) {
    const int j  = blockIdx.x * 256 + threadIdx.x;
    const int z  = blockIdx.y;
    const int r0 = z * (NN / 16);

    float m = -3.402823466e+38f;
#pragma unroll 8
    for (int r = 0; r < NN / 16; r++) m = fmaxf(m, P[(size_t)(r0 + r) * MM + j]);
    float s = 0.0f;
#pragma unroll 4
    for (int r = 0; r < NN / 16; r++) s += __expf(P[(size_t)(r0 + r) * MM + j] - m);
    pMax[z * MM + j] = m;
    pSum[z * MM + j] = s;
}

__global__ __launch_bounds__(256) void col_combine_kernel(const float* __restrict__ pMax,
                                                          const float* __restrict__ pSum,
                                                          float* __restrict__ colMax,
                                                          float* __restrict__ colRcp) {
    const int j = blockIdx.x * 256 + threadIdx.x;
    float m = -3.402823466e+38f;
#pragma unroll
    for (int z = 0; z < 16; z++) m = fmaxf(m, pMax[z * MM + j]);
    float s = 0.0f;
#pragma unroll
    for (int z = 0; z < 16; z++) s += pSum[z * MM + j] * __expf(pMax[z * MM + j] - m);
    colMax[j] = m;
    colRcp[j] = 1.0f / s;
}

// ---------------------------------------------------------------------------
// K5: fused softmax pass.  One read of p produces:
//   ab[i,k]  = exp(p - rowMax[i]) * rowRcp[i]        (bf16)
//   C [i,k]  = exp(p - colMax[k]) * colRcp[k]        (bf16)  == ba^T
//   visitSum[k] += sum_i ab[i,k]                     (f32 atomics)
// Block: 256 cols x 64 rows.
// ---------------------------------------------------------------------------
__global__ __launch_bounds__(256) void softmax_kernel(const float* __restrict__ P,
                                                      const float* __restrict__ rowMax,
                                                      const float* __restrict__ rowRcp,
                                                      const float* __restrict__ colMax,
                                                      const float* __restrict__ colRcp,
                                                      unsigned short* __restrict__ ab,
                                                      unsigned short* __restrict__ cm,
                                                      float* __restrict__ visitSum) {
    const int j  = blockIdx.x * 256 + threadIdx.x;
    const int i0 = blockIdx.y * 64;
    __shared__ float rm[64], rr[64];
    if (threadIdx.x < 64) {
        rm[threadIdx.x] = rowMax[i0 + threadIdx.x];
        rr[threadIdx.x] = rowRcp[i0 + threadIdx.x];
    }
    __syncthreads();

    const float cmx = colMax[j];
    const float crc = colRcp[j];
    float acc = 0.0f;
#pragma unroll 4
    for (int r = 0; r < 64; r++) {
        size_t idx = (size_t)(i0 + r) * MM + j;
        float x  = P[idx];
        float e1 = __expf(x - rm[r]) * rr[r];
        float e2 = __expf(x - cmx) * crc;
        ab[idx] = f32_to_bf16(e1);
        cm[idx] = f32_to_bf16(e2);
        acc += e1;
    }
    atomicAdd(&visitSum[j], acc);
}

// ---------------------------------------------------------------------------
// K6: visit loss -> out[1]
// ---------------------------------------------------------------------------
__global__ __launch_bounds__(256) void visit_kernel(const float* __restrict__ visitSum,
                                                    float* __restrict__ out) {
    __shared__ float red[256];
    const int t = threadIdx.x;
    float s = 0.0f;
    for (int j = t; j < MM; j += 256)
        s += __logf(EPS + visitSum[j] * (1.0f / (float)NN));
    red[t] = s; __syncthreads();
    for (int k = 128; k > 0; k >>= 1) { if (t < k) red[t] += red[t + k]; __syncthreads(); }
    if (t == 0) out[1] = -red[0] * (1.0f / (float)MM);
}

// ---------------------------------------------------------------------------
// K7: label bucketing -> counts[10], offsets[10], labelIdx[8192]
// ---------------------------------------------------------------------------
__global__ __launch_bounds__(256) void label_prep_kernel(const long long* __restrict__ labels,
                                                         int* __restrict__ labelIdx,
                                                         int* __restrict__ countsG,
                                                         int* __restrict__ offsG) {
    __shared__ int cnt[NLAB], off[NLAB], cur[NLAB];
    const int t = threadIdx.x;
    if (t < NLAB) cnt[t] = 0;
    __syncthreads();
    for (int i = t; i < NN; i += 256) atomicAdd(&cnt[(int)labels[i]], 1);
    __syncthreads();
    if (t == 0) {
        int a = 0;
        for (int c = 0; c < NLAB; c++) { off[c] = a; a += cnt[c]; }
    }
    __syncthreads();
    if (t < NLAB) { countsG[t] = cnt[t]; offsG[t] = off[t]; cur[t] = off[t]; }
    __syncthreads();
    for (int i = t; i < NN; i += 256) {
        int pos = atomicAdd(&cur[(int)labels[i]], 1);
        labelIdx[pos] = i;
    }
}

// ---------------------------------------------------------------------------
// K8: walker loss.  Per class c: aba_c = ab[rows_c,:] @ C[rows_c,:]^T via
// gathered-row 64x64 WMMA tiles over K=8192; epilogue accumulates
// -(1/(N*n_c)) * log(EPS + aba) for valid (i,j), atomicAdd into out[0].
// 8 waves: 4 (M) x 2 (N); each wave owns 16 rows x 32 cols = 2 WMMA tiles.
// ---------------------------------------------------------------------------
__global__ __launch_bounds__(256) void walker_kernel(const unsigned short* __restrict__ ab,
                                                     const unsigned short* __restrict__ cm,
                                                     const int* __restrict__ countsG,
                                                     const int* __restrict__ offsG,
                                                     const int* __restrict__ labelIdx,
                                                     float* __restrict__ out) {
    const int c   = blockIdx.z;
    const int n_c = countsG[c];
    const int iBase = blockIdx.x * 64;
    const int jBase = blockIdx.y * 64;
    if (iBase >= n_c || jBase >= n_c) return;  // block-uniform: EXEC stays full
    const int off = offsG[c];

    __shared__ int idxA[64], idxB[64];
    __shared__ __align__(16) unsigned short As[64 * PLDT];
    __shared__ __align__(16) unsigned short Bs[64 * PLDT];
    __shared__ float red[256];

    const int t = threadIdx.x;
    if (t < 64) {
        int r = iBase + t;
        idxA[t] = labelIdx[off + (r < n_c ? r : 0)];
    } else if (t < 128) {
        int u = t - 64;
        int r = jBase + u;
        idxB[u] = labelIdx[off + (r < n_c ? r : 0)];
    }
    __syncthreads();

    const int lane = t & 31;
    const int wave = t >> 5;
    const int wm   = wave & 3;   // 4 wave-rows of 16
    const int wn   = wave >> 2;  // 2 wave-cols of 32
    const int half = lane >> 4;
    const int l15  = lane & 15;

    v8f acc0 = {}, acc1 = {};

    for (int k0 = 0; k0 < MM; k0 += 64) {
#pragma unroll
        for (int s = t; s < 512; s += 256) {
            int r  = s >> 3;
            int ch = s & 7;
            *(uint4*)(&As[r * PLDT + ch * 8]) =
                *(const uint4*)(&ab[(size_t)idxA[r] * MM + k0 + ch * 8]);
            *(uint4*)(&Bs[r * PLDT + ch * 8]) =
                *(const uint4*)(&cm[(size_t)idxB[r] * MM + k0 + ch * 8]);
        }
        __syncthreads();

#pragma unroll
        for (int ks = 0; ks < 64; ks += 32) {
            FragBf aF, b0, b1;
            const unsigned short* rpA = &As[(wm * 16 + l15) * PLDT + ks];
            aF.q[0] = *(const uint4*)(rpA + 16 * half);
            aF.q[1] = *(const uint4*)(rpA + 16 * half + 8);
            const unsigned short* rp0 = &Bs[(wn * 32 + l15) * PLDT + ks];
            b0.q[0] = *(const uint4*)(rp0 + 16 * half);
            b0.q[1] = *(const uint4*)(rp0 + 16 * half + 8);
            const unsigned short* rp1 = &Bs[(wn * 32 + 16 + l15) * PLDT + ks];
            b1.q[0] = *(const uint4*)(rp1 + 16 * half);
            b1.q[1] = *(const uint4*)(rp1 + 16 * half + 8);
            acc0 = wmma_bf16(aF, b0, acc0);
            acc1 = wmma_bf16(aF, b1, acc1);
        }
        __syncthreads();
    }

    // epilogue: masked log-accumulate
    float local = 0.0f;
    const int il0 = iBase + wm * 16 + 8 * half;
    const int jl0 = jBase + wn * 32 + l15;
    const int jl1 = jl0 + 16;
#pragma unroll
    for (int r = 0; r < 8; r++) {
        int i = il0 + r;
        if (i < n_c && jl0 < n_c) local += __logf(EPS + acc0[r]);
        if (i < n_c && jl1 < n_c) local += __logf(EPS + acc1[r]);
    }
    red[t] = local; __syncthreads();
    for (int s = 128; s > 0; s >>= 1) { if (t < s) red[t] += red[t + s]; __syncthreads(); }
    if (t == 0) atomicAdd(out, -red[0] / ((float)NN * (float)n_c));
}

// ---------------------------------------------------------------------------
// launcher
// ---------------------------------------------------------------------------
extern "C" void kernel_launch(void* const* d_in, const int* in_sizes, int n_in,
                              void* d_out, int out_size, void* d_ws, size_t ws_size,
                              hipStream_t stream) {
    const float*     a      = (const float*)d_in[0];
    const float*     b      = (const float*)d_in[1];
    const long long* labels = (const long long*)d_in[2];
    float* out = (float*)d_out;
    char*  ws  = (char*)d_ws;

    const size_t P_BYTES  = (size_t)NN * MM * 4;  // 256 MB
    const size_t AB_BYTES = (size_t)NN * MM * 2;  // 128 MB each

    float*          P       = (float*)(ws);
    unsigned short* ab      = (unsigned short*)(ws + P_BYTES);
    unsigned short* cm      = (unsigned short*)(ws + P_BYTES + AB_BYTES);
    size_t o = P_BYTES + 2 * AB_BYTES;
    float* rowMax   = (float*)(ws + o); o += (size_t)NN * 4;
    float* rowRcp   = (float*)(ws + o); o += (size_t)NN * 4;
    float* colMax   = (float*)(ws + o); o += (size_t)MM * 4;
    float* colRcp   = (float*)(ws + o); o += (size_t)MM * 4;
    float* visitSum = (float*)(ws + o); o += (size_t)MM * 4;
    float* colPMax  = (float*)(ws + o); o += (size_t)16 * MM * 4;
    float* colPSum  = (float*)(ws + o); o += (size_t)16 * MM * 4;
    int*   labelIdx = (int*)(ws + o);   o += (size_t)NN * 4;
    int*   countsG  = (int*)(ws + o);   o += 64;
    int*   offsG    = (int*)(ws + o);   o += 64;

    init_kernel<<<dim3(MM / 256), 256, 0, stream>>>(out, visitSum);
    gemm_p_kernel<<<dim3(MM / 128, NN / 128), 256, 0, stream>>>(a, b, P);
    row_stats_kernel<<<dim3(NN), 256, 0, stream>>>(P, rowMax, rowRcp);
    col_part_kernel<<<dim3(MM / 256, 16), 256, 0, stream>>>(P, colPMax, colPSum);
    col_combine_kernel<<<dim3(MM / 256), 256, 0, stream>>>(colPMax, colPSum, colMax, colRcp);
    softmax_kernel<<<dim3(MM / 256, NN / 64), 256, 0, stream>>>(P, rowMax, rowRcp, colMax,
                                                                colRcp, ab, cm, visitSum);
    visit_kernel<<<dim3(1), 256, 0, stream>>>(visitSum, out);
    label_prep_kernel<<<dim3(1), 256, 0, stream>>>(labels, labelIdx, countsG, offsG);
    walker_kernel<<<dim3(NN / 64, NN / 64, NLAB), 256, 0, stream>>>(ab, cm, countsG, offsG,
                                                                    labelIdx, out);
}
